// Model_26053271617687
// MI455X (gfx1250) — compile-verified, matrix operand order checked by reading, not run on hardware
//
#include <hip/hip_runtime.h>
#include <hip/hip_bf16.h>

// ---------------------------------------------------------------------------
// Seq2seq LSTM + Bahdanau attention for MI455X (gfx1250, wave32, WMMA).
// All GEMMs run through v_wmma_f32_16x16x32_bf16 with fp32 accumulation.
// Weights converted once per launch to transposed bf16; activations bf16
// end-to-end. Each wave computes NT 16x16 tiles sharing one A-fragment per
// K-step (NT independent WMMA chains), with global_prefetch lookahead on the
// streamed weights.
// ---------------------------------------------------------------------------

typedef __bf16 bf16_t;
typedef __attribute__((ext_vector_type(16))) __bf16 v16bf;
typedef __attribute__((ext_vector_type(8)))  __bf16 v8bf;
typedef __attribute__((ext_vector_type(8)))  float  v8f;

#define BATCH 32
#define MAXLEN 64
#define EMBED 512
#define DH 1024          // DEC_H
#define G4 4096          // 4*DEC_H
#define VOCAB 32000

__device__ __forceinline__ bf16_t f2bf(float f) {
  union { float f; unsigned u; } v; v.f = f;
  unsigned r = (v.u + 0x7FFFu + ((v.u >> 16) & 1u)) >> 16;  // RNE
  unsigned short s = (unsigned short)r;
  bf16_t b; __builtin_memcpy(&b, &s, 2); return b;
}
__device__ __forceinline__ float sigmoidf_(float x) { return 1.0f / (1.0f + __expf(-x)); }

// ---------------------------------------------------------------------------
// Weight prep: fp32 W[K][N] -> bf16 WT[N][K] (transposed for contiguous frags)
// ---------------------------------------------------------------------------
__global__ void k_cvt_T(const float* __restrict__ W, bf16_t* __restrict__ WT,
                        int K, int N) {
  long idx = (long)blockIdx.x * blockDim.x + threadIdx.x;
  long total = (long)K * N;
  if (idx >= total) return;
  long n = idx / K, k = idx - n * K;
  WT[idx] = f2bf(W[k * (long)N + n]);
}

__global__ void k_zero(float* p, int n) {
  int i = blockIdx.x * blockDim.x + threadIdx.x;
  if (i < n) p[i] = 0.0f;
}

__global__ void k_zero_bf(bf16_t* p, int n) {
  int i = blockIdx.x * blockDim.x + threadIdx.x;
  if (i < n) p[i] = f2bf(0.0f);
}

__global__ void k_set_tok(int* tok, int val) {
  int i = threadIdx.x;
  if (i < BATCH) tok[i] = val;
}

// gather embedding rows to bf16: out[b][d] = bf16(emb[toks[b*tstride]][d])
__global__ void k_gather(const float* __restrict__ emb, int D,
                         const int* __restrict__ toks, int tstride,
                         bf16_t* __restrict__ out) {
  int idx = blockIdx.x * blockDim.x + threadIdx.x;
  int b = idx / D, d = idx - b * D;
  out[idx] = f2bf(emb[(long)toks[b * tstride] * D + d]);
}

// ---------------------------------------------------------------------------
// WMMA GEMM: OUT(32 x N) = X(32 x K1) @ W1T' + H(32 x K2) @ W2T' + bias
// X/H/W1T/W2T all bf16; W*T transposed [N][ld].
// Block = 256 threads = 8 waves; wave w -> (m_tile = w&1, n_group = w>>1).
// Each wave computes NT 16x16 tiles sharing one A-fragment per K-step,
// giving NT independent WMMA chains. Block covers 32 x (64*NT).
// act: 0 = none, 1 = relu. Writes fp32 OUT32 and/or bf16 OUT16.
// ---------------------------------------------------------------------------
__device__ __forceinline__ v16bf load_frag(const bf16_t* __restrict__ rowp,
                                           int kb, int half) {
  // A (16x32 bf16): lane half 0 -> K {0-7, 16-23}, half 1 -> {8-15, 24-31}
  // B (32x16) mirrored per lane half; both are two contiguous 16B loads.
  v8bf lo = *(const v8bf*)(rowp + kb + half * 8);
  v8bf hi = *(const v8bf*)(rowp + kb + 16 + half * 8);
  v16bf f;
#pragma unroll
  for (int i = 0; i < 8; ++i) f[i] = lo[i];
#pragma unroll
  for (int i = 0; i < 8; ++i) f[8 + i] = hi[i];
  return f;
}

template <int NT>
__global__ void k_wmma_gemm(const bf16_t* __restrict__ X, int K1, int ldx,
                            const bf16_t* __restrict__ W1T, int ldw1,
                            const bf16_t* __restrict__ H, int K2, int ldh,
                            const bf16_t* __restrict__ W2T, int ldw2,
                            const float* __restrict__ bias,
                            float* __restrict__ OUT32,
                            bf16_t* __restrict__ OUT16, int N, int act) {
  const int wave = threadIdx.x >> 5;
  const int lane = threadIdx.x & 31;
  const int half = lane >> 4;
  const int l16  = lane & 15;
  const int m_tile = wave & 1;                             // rows 0-15 / 16-31
  const int n_base = blockIdx.x * (64 * NT) + (wave >> 1) * (16 * NT);
  const int m = m_tile * 16 + l16;                         // A row for this lane

  v8f zero = {0.f, 0.f, 0.f, 0.f, 0.f, 0.f, 0.f, 0.f};
  v8f acc[NT];
#pragma unroll
  for (int j = 0; j < NT; ++j) acc[j] = zero;

  const bf16_t* xr = X + (long)m * ldx;
  {
    const bf16_t* wr = W1T + (long)(n_base + l16) * ldw1;
    for (int kb = 0; kb < K1; kb += 32) {
      v16bf a = load_frag(xr, kb, half);
#pragma unroll
      for (int j = 0; j < NT; ++j) {
        const bf16_t* wj = wr + (long)j * 16 * ldw1;
        __builtin_prefetch(wj + kb + 128, 0, 1);           // global_prefetch_b8
        v16bf b = load_frag(wj, kb, half);
        acc[j] = __builtin_amdgcn_wmma_f32_16x16x32_bf16(false, a, false, b,
                                                         (short)0, acc[j],
                                                         false, false);
      }
    }
  }
  if (K2 > 0) {
    const bf16_t* hr = H + (long)m * ldh;
    const bf16_t* w2 = W2T + (long)(n_base + l16) * ldw2;
    for (int kb = 0; kb < K2; kb += 32) {
      v16bf a = load_frag(hr, kb, half);
#pragma unroll
      for (int j = 0; j < NT; ++j) {
        const bf16_t* wj = w2 + (long)j * 16 * ldw2;
        __builtin_prefetch(wj + kb + 128, 0, 1);
        v16bf b = load_frag(wj, kb, half);
        acc[j] = __builtin_amdgcn_wmma_f32_16x16x32_bf16(false, a, false, b,
                                                         (short)0, acc[j],
                                                         false, false);
      }
    }
  }

#pragma unroll
  for (int j = 0; j < NT; ++j) {
    int n = n_base + j * 16 + l16;
    float bv = bias ? bias[n] : 0.0f;
#pragma unroll
    for (int v = 0; v < 8; ++v) {                          // C/D: row = v + 8*half
      int row = m_tile * 16 + v + half * 8;
      float val = acc[j][v] + bv;
      if (act == 1) val = fmaxf(val, 0.0f);
      long o = (long)row * N + n;
      if (OUT32) OUT32[o] = val;
      if (OUT16) OUT16[o] = f2bf(val);
    }
  }
}

// ---------------------------------------------------------------------------
// LSTM pointwise: gates(32x4096, order i,f,g,o) + c -> c, bf16 h
// optional enc_slot[b*65536 + d] = h (fp32, enc_outs [B][L][DH], slot = +t*DH)
// ---------------------------------------------------------------------------
__global__ void k_lstm_point(const float* __restrict__ g,
                             bf16_t* __restrict__ h_bf, float* __restrict__ c,
                             float* __restrict__ enc_slot) {
  int idx = blockIdx.x * blockDim.x + threadIdx.x;   // 32*1024
  int b = idx >> 10, d = idx & 1023;
  const float* gr = g + b * G4;
  float gi = sigmoidf_(gr[d]);
  float gf = sigmoidf_(gr[DH + d]);
  float gg = tanhf(gr[2 * DH + d]);
  float go = sigmoidf_(gr[3 * DH + d]);
  float cn = gf * c[idx] + gi * gg;
  float hn = go * tanhf(cn);
  c[idx] = cn;
  h_bf[idx] = f2bf(hn);
  if (enc_slot) enc_slot[b * (MAXLEN * DH) + d] = hn;
}

// softmax over 64 attention scores, in place. grid=32, block=64
__global__ void k_softmax64(float* __restrict__ s) {
  __shared__ float red[64];
  int b = blockIdx.x, t = threadIdx.x;
  float v = s[b * 64 + t];
  red[t] = v; __syncthreads();
  for (int off = 32; off > 0; off >>= 1) {
    if (t < off) red[t] = fmaxf(red[t], red[t + off]);
    __syncthreads();
  }
  float mx = red[0]; __syncthreads();
  float e = __expf(v - mx);
  red[t] = e; __syncthreads();
  for (int off = 32; off > 0; off >>= 1) {
    if (t < off) red[t] += red[t + off];
    __syncthreads();
  }
  s[b * 64 + t] = e / red[0];
}

// ctx[b][d] = bf16( sum_l aw[b][l] * enc_outs[b][l][d] )
__global__ void k_ctx(const float* __restrict__ aw,
                      const float* __restrict__ eo, bf16_t* __restrict__ ctx) {
  int idx = blockIdx.x * blockDim.x + threadIdx.x;   // 32*1024
  int b = idx >> 10, d = idx & 1023;
  const float* a = aw + b * 64;
  const float* e = eo + (long)b * (MAXLEN * DH) + d;
  float s = 0.0f;
#pragma unroll 4
  for (int l = 0; l < MAXLEN; ++l) s += a[l] * e[l * DH];
  ctx[idx] = f2bf(s);
}

// log-softmax over 32000 logits per batch row + greedy argmax -> tok
// grid=32, block=256; outp indexes [b*(64*32000) + v] (t offset folded in)
__global__ void k_logsoftmax(const float* __restrict__ logits,
                             float* __restrict__ outp, int* __restrict__ tok) {
  __shared__ float sm[256];
  __shared__ int   si[256];
  int b = blockIdx.x, t = threadIdx.x;
  const float* row = logits + (long)b * VOCAB;
  float mx = -3.4e38f; int mi = 0;
  for (int v = t; v < VOCAB; v += 256) {
    float x = row[v];
    if (x > mx) { mx = x; mi = v; }
  }
  sm[t] = mx; si[t] = mi; __syncthreads();
  for (int off = 128; off > 0; off >>= 1) {
    if (t < off) {
      if (sm[t + off] > sm[t] || (sm[t + off] == sm[t] && si[t + off] < si[t])) {
        sm[t] = sm[t + off]; si[t] = si[t + off];
      }
    }
    __syncthreads();
  }
  float m0 = sm[0]; int amax = si[0]; __syncthreads();
  float s = 0.0f;
  for (int v = t; v < VOCAB; v += 256) s += __expf(row[v] - m0);
  sm[t] = s; __syncthreads();
  for (int off = 128; off > 0; off >>= 1) {
    if (t < off) sm[t] += sm[t + off];
    __syncthreads();
  }
  float logZ = m0 + __logf(sm[0]);
  float* orow = outp + (long)b * (MAXLEN * VOCAB);
  for (int v = t; v < VOCAB; v += 256) orow[v] = row[v] - logZ;
  if (t == 0) tok[b] = amax;
}

// ---------------------------------------------------------------------------
extern "C" void kernel_launch(void* const* d_in, const int* in_sizes, int n_in,
                              void* d_out, int out_size, void* d_ws, size_t ws_size,
                              hipStream_t stream) {
  (void)in_sizes; (void)n_in; (void)out_size; (void)ws_size;
  const int*   x         = (const int*)  d_in[0];   // (32,64)
  const float* enc_embed = (const float*)d_in[1];   // (32000,512)
  const float* enc_wx    = (const float*)d_in[2];   // (512,4096)
  const float* enc_wh    = (const float*)d_in[3];   // (1024,4096)
  const float* enc_b     = (const float*)d_in[4];
  const float* dec_embed = (const float*)d_in[5];   // (32000,1024)
  const float* attn_w    = (const float*)d_in[6];   // (2048,64)
  const float* attn_b    = (const float*)d_in[7];
  const float* comb_w    = (const float*)d_in[8];   // (2048,1024)
  const float* comb_b    = (const float*)d_in[9];
  const float* dec_wx    = (const float*)d_in[10];  // (1024,4096)
  const float* dec_wh    = (const float*)d_in[11];  // (1024,4096)
  const float* dec_b     = (const float*)d_in[12];
  const float* out_w     = (const float*)d_in[13];  // (1024,32000)
  const float* out_b     = (const float*)d_in[14];
  float* out = (float*)d_out;                       // (32,64,32000)

  // ---- workspace carve-out (256B aligned) ----
  char* w = (char*)d_ws;
  auto take = [&](size_t bytes) -> void* {
    void* p = (void*)w; w += (bytes + 255) & ~(size_t)255; return p;
  };
  bf16_t* enc_wxT  = (bf16_t*)take((size_t)G4 * EMBED * 2);
  bf16_t* enc_whT  = (bf16_t*)take((size_t)G4 * DH * 2);
  bf16_t* dec_wxT  = (bf16_t*)take((size_t)G4 * DH * 2);
  bf16_t* dec_whT  = (bf16_t*)take((size_t)G4 * DH * 2);
  bf16_t* comb_wT  = (bf16_t*)take((size_t)DH * 2048 * 2);
  bf16_t* attn_wT  = (bf16_t*)take((size_t)64 * 2048 * 2);
  bf16_t* out_wT   = (bf16_t*)take((size_t)VOCAB * DH * 2);
  float*  enc_outs = (float*)take((size_t)BATCH * MAXLEN * DH * 4);
  float*  c_enc    = (float*)take((size_t)BATCH * DH * 4);
  float*  c_dec    = (float*)take((size_t)BATCH * DH * 4);
  bf16_t* h_enc_bf = (bf16_t*)take((size_t)BATCH * DH * 2);
  bf16_t* h_dec_bf = (bf16_t*)take((size_t)BATCH * DH * 2);
  bf16_t* e_bf     = (bf16_t*)take((size_t)BATCH * DH * 2);
  bf16_t* ctx_bf   = (bf16_t*)take((size_t)BATCH * DH * 2);
  bf16_t* inp_bf   = (bf16_t*)take((size_t)BATCH * DH * 2);
  float*  gates    = (float*)take((size_t)BATCH * G4 * 4);
  float*  scores   = (float*)take((size_t)BATCH * MAXLEN * 4);
  float*  logits   = (float*)take((size_t)BATCH * VOCAB * 4);
  int*    tok      = (int*)take(256);

  const int TPB = 256;
  auto cvt = [&](const float* W, bf16_t* WT, long K, long N) {
    long total = K * N;
    k_cvt_T<<<(unsigned)((total + TPB - 1) / TPB), TPB, 0, stream>>>(W, WT, (int)K, (int)N);
  };

  // ---- weight prep (bf16, transposed) ----
  cvt(enc_wx, enc_wxT, EMBED, G4);
  cvt(enc_wh, enc_whT, DH, G4);
  cvt(dec_wx, dec_wxT, DH, G4);
  cvt(dec_wh, dec_whT, DH, G4);
  cvt(comb_w, comb_wT, 2048, DH);
  cvt(attn_w, attn_wT, 2048, 64);
  cvt(out_w,  out_wT,  DH, VOCAB);

  const int NHD = BATCH * DH;            // 32768
  k_zero<<<NHD / TPB, TPB, 0, stream>>>(c_enc, NHD);
  k_zero<<<NHD / TPB, TPB, 0, stream>>>(c_dec, NHD);
  k_zero_bf<<<NHD / TPB, TPB, 0, stream>>>(h_enc_bf, NHD);
  k_zero_bf<<<NHD / TPB, TPB, 0, stream>>>(h_dec_bf, NHD);
  k_set_tok<<<1, 32, 0, stream>>>(tok, 127);

  // ---- encoder: 64 serial steps ----
  for (int t = 0; t < MAXLEN; ++t) {
    k_gather<<<(BATCH * EMBED) / TPB, TPB, 0, stream>>>(enc_embed, EMBED, x + t, MAXLEN, e_bf);
    // gates = e @ enc_wx + h @ enc_wh + b   (NT=2: 32 blocks)
    k_wmma_gemm<2><<<G4 / 128, TPB, 0, stream>>>(e_bf, EMBED, EMBED, enc_wxT, EMBED,
                                                 h_enc_bf, DH, DH, enc_whT, DH,
                                                 enc_b, gates, nullptr, G4, 0);
    k_lstm_point<<<NHD / TPB, TPB, 0, stream>>>(gates, h_enc_bf, c_enc, enc_outs + t * DH);
  }

  // ---- decoder: 64 serial steps with greedy feedback ----
  for (int t = 0; t < MAXLEN; ++t) {
    k_gather<<<(BATCH * DH) / TPB, TPB, 0, stream>>>(dec_embed, DH, tok, 1, e_bf);
    // scores = [e, h] @ attn_w + attn_b  (two K-segments into transposed attn_w)
    k_wmma_gemm<1><<<1, TPB, 0, stream>>>(e_bf, DH, DH, attn_wT, 2048,
                                          h_dec_bf, DH, DH, attn_wT + DH, 2048,
                                          attn_b, scores, nullptr, 64, 0);
    k_softmax64<<<BATCH, 64, 0, stream>>>(scores);
    k_ctx<<<NHD / TPB, TPB, 0, stream>>>(scores, enc_outs, ctx_bf);
    // inp = relu([e, ctx] @ comb_w + comb_b) -> bf16 directly (NT=2: 8 blocks)
    k_wmma_gemm<2><<<DH / 128, TPB, 0, stream>>>(e_bf, DH, DH, comb_wT, 2048,
                                                 ctx_bf, DH, DH, comb_wT + DH, 2048,
                                                 comb_b, nullptr, inp_bf, DH, 1);
    // gates = inp @ dec_wx + h @ dec_wh + b  (NT=2: 32 blocks)
    k_wmma_gemm<2><<<G4 / 128, TPB, 0, stream>>>(inp_bf, DH, DH, dec_wxT, DH,
                                                 h_dec_bf, DH, DH, dec_whT, DH,
                                                 dec_b, gates, nullptr, G4, 0);
    k_lstm_point<<<NHD / TPB, TPB, 0, stream>>>(gates, h_dec_bf, c_dec, nullptr);
    // logits = h @ out_w + out_b  (dominant GEMM; NT=4: 125 blocks, A reuse x4)
    k_wmma_gemm<4><<<VOCAB / 256, TPB, 0, stream>>>(h_dec_bf, DH, DH, out_wT, DH,
                                                    nullptr, 0, 0, nullptr, 0,
                                                    out_b, logits, nullptr, VOCAB, 0);
    k_logsoftmax<<<BATCH, TPB, 0, stream>>>(logits, out + (long)t * VOCAB, tok);
  }
}
